// EfficientNet_my_28741921145195
// MI455X (gfx1250) — 8-bit matrix operand order repaired and checked by reading, compiled, not run on hardware
//
#include <hip/hip_runtime.h>
#include <hip/hip_bf16.h>
#include <cstdint>
#include <cstddef>


typedef __attribute__((ext_vector_type(8))) int v8i;
typedef __attribute__((ext_vector_type(4))) int v4i;

static inline int pad16(int x) { return (x + 15) & ~15; }
static inline int pad32i(int x) { return (x + 31) & ~31; }
static inline int pad64(int x) { return (x + 63) & ~63; }


__device__ __forceinline__ v8i load_a_frag(const int8_t* __restrict__ A,
                                           size_t arow, int k0, int half)
{
  v8i r;
#pragma unroll
  for (int v = 0; v < 8; ++v) {
    const int ka = (v & 3) * 4 + (v >> 2) * 32 + half * 16;
    r[v] = *(const int*)(A + arow + k0 + ka);
  }
  return r;
}

__device__ __forceinline__ v8i load_b_frag(const int8_t* shB, int base, int half)
{
  const v4i lo = *(const v4i*)(shB + base + half * 16);
  const v4i hi = *(const v4i*)(shB + base + 32 + half * 16);
  v8i r;
  r[0] = lo[0]; r[1] = lo[1]; r[2] = lo[2]; r[3] = lo[3];
  r[4] = hi[0]; r[5] = hi[1]; r[6] = hi[2]; r[7] = hi[3];
  return r;
}

__device__ __forceinline__ v8i wmma_iu8(v8i a, v8i b, v8i c)
{
  return __builtin_amdgcn_wmma_i32_16x16x64_iu8(true, a, true, b, c,
                                                false, false);
}

__global__ __launch_bounds__(128) void k_bingemm(
    const int8_t* __restrict__ A, const int8_t* __restrict__ B,
    float* __restrict__ C, const float* __restrict__ bias,
    int M, int N, int Kpad, int Mtiles)
{
  extern __shared__ int8_t shB[];
  const int tid  = threadIdx.x;
  const int lane = tid & 31;
  const int wave = tid >> 5;
  const int tm   = blockIdx.x * 4 + wave;
  const int tn2  = blockIdx.y;

  {
    const int total = 32 * Kpad;
    const int8_t* g = B + (size_t)tn2 * 32 * (size_t)Kpad;
    for (int o = tid * 16; o < total; o += 128 * 16) {
      const unsigned l = (unsigned)(uintptr_t)(shB + o);
      const int8_t* gp = g + o;
      asm volatile("global_load_async_to_lds_b128 %0, %1, off"
                   :: "v"(l), "v"(gp) : "memory");
    }
    asm volatile("s_wait_asynccnt 0x0" ::: "memory");
  }
  __syncthreads();

  if (tm < Mtiles) {
    const int half = lane >> 4;
    const int l16  = lane & 15;
    const size_t arow = (size_t)(tm * 16 + l16) * (size_t)Kpad;
    const int brow0 = l16 * Kpad;
    const int brow1 = (16 + l16) * Kpad;

    v8i acc0 = {0, 0, 0, 0, 0, 0, 0, 0};
    v8i acc1 = {0, 0, 0, 0, 0, 0, 0, 0};

    v8i a0 = load_a_frag(A, arow, 0, half);
    int k0 = 0;
    while (k0 + 128 <= Kpad) {
      v8i a1 = load_a_frag(A, arow, k0 + 64, half);
      acc0 = wmma_iu8(a0, load_b_frag(shB, brow0 + k0, half), acc0);
      acc1 = wmma_iu8(a0, load_b_frag(shB, brow1 + k0, half), acc1);
      if (k0 + 128 < Kpad) a0 = load_a_frag(A, arow, k0 + 128, half);
      acc0 = wmma_iu8(a1, load_b_frag(shB, brow0 + k0 + 64, half), acc0);
      acc1 = wmma_iu8(a1, load_b_frag(shB, brow1 + k0 + 64, half), acc1);
      k0 += 128;
    }
    if (k0 < Kpad) {
      acc0 = wmma_iu8(a0, load_b_frag(shB, brow0 + k0, half), acc0);
      acc1 = wmma_iu8(a0, load_b_frag(shB, brow1 + k0, half), acc1);
    }

#pragma unroll
    for (int r = 0; r < 8; ++r) {
      const int m = tm * 16 + half * 8 + r;
      if (m < M) {
        const int n0 = tn2 * 32 + l16;
        const int n1 = n0 + 16;
        if (n0 < N) {
          float f = (float)acc0[r];
          if (bias) f += bias[n0];
          C[(size_t)m * N + n0] = f;
        }
        if (n1 < N) {
          float f = (float)acc1[r];
          if (bias) f += bias[n1];
          C[(size_t)m * N + n1] = f;
        }
      }
    }
  }
}


__global__ void k_stem(const float* __restrict__ x, const float* __restrict__ w,
                       float* __restrict__ out)
{
  const size_t idx = (size_t)blockIdx.x * blockDim.x + threadIdx.x;
  const size_t tot = (size_t)32 * 112 * 112 * 32;
  if (idx >= tot) return;
  const int co = (int)(idx & 31);
  size_t m = idx >> 5;
  const int wo = (int)(m % 112); m /= 112;
  const int ho = (int)(m % 112);
  const int n  = (int)(m / 112);
  float acc = 0.f;
  for (int ci = 0; ci < 3; ++ci)
    for (int kh = 0; kh < 3; ++kh) {
      const int hi = ho * 2 - 1 + kh;
      if ((unsigned)hi >= 224u) continue;
      for (int kw = 0; kw < 3; ++kw) {
        const int wi = wo * 2 - 1 + kw;
        if ((unsigned)wi >= 224u) continue;
        const float wv = w[((co * 3 + ci) * 3 + kh) * 3 + kw];
        const float sw = (wv > 0.f) ? 1.f : (wv < 0.f ? -1.f : 0.f);
        acc += x[(((size_t)n * 3 + ci) * 224 + hi) * 224 + wi] * sw;
      }
    }
  out[idx] = acc;
}

__global__ void k_bn_stats(const float* __restrict__ x, float* __restrict__ st,
                           int M, int C, float eps)
{
  const int c = blockIdx.x;
  float s = 0.f, s2 = 0.f;
  for (int m = threadIdx.x; m < M; m += blockDim.x) {
    const float v = x[(size_t)m * C + c];
    s += v; s2 += v * v;
  }
  __shared__ float sh0[256], sh1[256];
  sh0[threadIdx.x] = s; sh1[threadIdx.x] = s2;
  __syncthreads();
  for (int t = 128; t > 0; t >>= 1) {
    if ((int)threadIdx.x < t) {
      sh0[threadIdx.x] += sh0[threadIdx.x + t];
      sh1[threadIdx.x] += sh1[threadIdx.x + t];
    }
    __syncthreads();
  }
  if (threadIdx.x == 0) {
    const float mean = sh0[0] / (float)M;
    const float var  = sh1[0] / (float)M - mean * mean;
    st[c] = mean;
    st[C + c] = rsqrtf(var + eps);
  }
}

__global__ void k_bn_act(const float* __restrict__ x, const float* __restrict__ st,
                         const float* __restrict__ g, const float* __restrict__ b,
                         float* __restrict__ outf, int8_t* __restrict__ outs,
                         int M, int C, int Mp, int Kp, int flags)
{
  const size_t idx = (size_t)blockIdx.x * blockDim.x + threadIdx.x;
  const size_t tot = (size_t)Mp * Kp;
  if (idx >= tot) return;
  const int c = (int)(idx % Kp);
  const size_t m = idx / Kp;
  float y = 0.f;
  const bool in = (m < (size_t)M) && (c < C);
  if (in) {
    const float v = x[m * C + c];
    y = (v - st[c]) * st[C + c] * g[c] + b[c];
    if (flags & 1) y = fminf(1.f, fmaxf(-1.f, y));
    if (flags & 2) y = 1.f / (1.f + expf(-y));
    if (outf) outf[m * C + c] = y;
  }
  if (outs) outs[idx] = in ? (int8_t)((y > 0.f) - (y < 0.f)) : (int8_t)0;
}

__global__ void k_signpack(const float* __restrict__ x, int8_t* __restrict__ o,
                           int M, int C, int Mp, int Kp)
{
  const size_t idx = (size_t)blockIdx.x * blockDim.x + threadIdx.x;
  const size_t tot = (size_t)Mp * Kp;
  if (idx >= tot) return;
  const int c = (int)(idx % Kp);
  const size_t m = idx / Kp;
  int8_t s = 0;
  if (m < (size_t)M && c < C) {
    const float v = x[m * C + c];
    s = (int8_t)((v > 0.f) - (v < 0.f));
  }
  o[idx] = s;
}

__global__ void k_wpack(const float* __restrict__ w, int8_t* __restrict__ o,
                        int O, int I, int Op, int Kp)
{
  const size_t idx = (size_t)blockIdx.x * blockDim.x + threadIdx.x;
  const size_t tot = (size_t)Op * Kp;
  if (idx >= tot) return;
  const int i = (int)(idx % Kp);
  const int oo = (int)(idx / Kp);
  int8_t s = 0;
  if (oo < O && i < I) {
    const float v = w[(size_t)oo * I + i];
    s = (int8_t)((v > 0.f) - (v < 0.f));
  }
  o[idx] = s;
}

__global__ void k_sign8(const float* __restrict__ w, int8_t* __restrict__ o, int n)
{
  const int i = blockIdx.x * blockDim.x + threadIdx.x;
  if (i >= n) return;
  const float v = w[i];
  o[i] = (int8_t)((v > 0.f) - (v < 0.f));
}

__global__ void k_dwconv(const int8_t* __restrict__ in, const int8_t* __restrict__ w,
                         float* __restrict__ out, int N, int H, int W, int C,
                         int Kp, int k, int s, int p, int Ho, int Wo)
{
  const size_t idx = (size_t)blockIdx.x * blockDim.x + threadIdx.x;
  const size_t tot = (size_t)N * Ho * Wo * C;
  if (idx >= tot) return;
  const int c = (int)(idx % C);
  size_t t = idx / C;
  const int wo = (int)(t % Wo); t /= Wo;
  const int ho = (int)(t % Ho);
  const int n  = (int)(t / Ho);
  int acc = 0;
  for (int kh = 0; kh < k; ++kh) {
    const int hi = ho * s - p + kh;
    if ((unsigned)hi >= (unsigned)H) continue;
    for (int kw = 0; kw < k; ++kw) {
      const int wi = wo * s - p + kw;
      if ((unsigned)wi >= (unsigned)W) continue;
      acc += (int)in[(((size_t)n * H + hi) * W + wi) * Kp + c] *
             (int)w[(c * k + kh) * k + kw];
    }
  }
  out[idx] = (float)acc;
}

__global__ void k_avgpool(const float* __restrict__ x, float* __restrict__ o,
                          int N, int HW, int C)
{
  const int idx = blockIdx.x * blockDim.x + threadIdx.x;
  if (idx >= N * C) return;
  const int c = idx % C, n = idx / C;
  const float* p = x + (size_t)n * HW * C + c;
  float s = 0.f;
  for (int i = 0; i < HW; ++i) s += p[(size_t)i * C];
  o[idx] = s / (float)HW;
}

__global__ void k_gate_sign(const float* __restrict__ x, const float* __restrict__ gate,
                            int8_t* __restrict__ o, int M, int C, int Mp, int Kp, int HW)
{
  const size_t idx = (size_t)blockIdx.x * blockDim.x + threadIdx.x;
  const size_t tot = (size_t)Mp * Kp;
  if (idx >= tot) return;
  const int c = (int)(idx % Kp);
  const size_t m = idx / Kp;
  int8_t s = 0;
  if (m < (size_t)M && c < C) {
    const float v = x[m * C + c] * gate[(m / HW) * C + c];
    s = (int8_t)((v > 0.f) - (v < 0.f));
  }
  o[idx] = s;
}

__global__ void k_add(const float* __restrict__ a, const float* __restrict__ b,
                      float* __restrict__ o, long long n)
{
  const long long i = (long long)blockIdx.x * blockDim.x + threadIdx.x;
  if (i < n) o[i] = a[i] + b[i];
}

__global__ void k_logsoftmax(const float* __restrict__ x, float* __restrict__ o, int C)
{
  const int n = blockIdx.x;
  const float* row = x + (size_t)n * C;
  __shared__ float sh[256];
  float mx = -3.4e38f;
  for (int c = threadIdx.x; c < C; c += blockDim.x) mx = fmaxf(mx, row[c]);
  sh[threadIdx.x] = mx; __syncthreads();
  for (int t = 128; t > 0; t >>= 1) {
    if ((int)threadIdx.x < t) sh[threadIdx.x] = fmaxf(sh[threadIdx.x], sh[threadIdx.x + t]);
    __syncthreads();
  }
  mx = sh[0]; __syncthreads();
  float s = 0.f;
  for (int c = threadIdx.x; c < C; c += blockDim.x) s += expf(row[c] - mx);
  sh[threadIdx.x] = s; __syncthreads();
  for (int t = 128; t > 0; t >>= 1) {
    if ((int)threadIdx.x < t) sh[threadIdx.x] += sh[threadIdx.x + t];
    __syncthreads();
  }
  const float lse = mx + logf(sh[0]);
  for (int c = threadIdx.x; c < C; c += blockDim.x)
    o[(size_t)n * C + c] = row[c] - lse;
}


struct Cfg { int ci, co, k, e, s; };
static const Cfg CFGS[16] = {
  {32, 16, 3, 1, 1},
  {16, 24, 3, 6, 2}, {24, 24, 3, 6, 1},
  {24, 40, 5, 6, 2}, {40, 40, 5, 6, 1},
  {40, 80, 3, 6, 2}, {80, 80, 3, 6, 1}, {80, 80, 3, 6, 1},
  {80, 112, 5, 6, 1}, {112, 112, 5, 6, 1}, {112, 112, 5, 6, 1},
  {112, 192, 5, 6, 2}, {192, 192, 5, 6, 1}, {192, 192, 5, 6, 1}, {192, 192, 5, 6, 1},
  {192, 320, 3, 6, 1},
};

#define MB ((size_t)1 << 20)
static const size_t F_X0    = 0;
static const size_t F_X1    = 64 * MB;
static const size_t F_GEMM  = 128 * MB;
static const size_t F_DWRAW = 288 * MB;
static const size_t F_DWACT = 352 * MB;
static const size_t F_TMP   = 416 * MB;
static const size_t S_IN    = 448 * MB;
static const size_t S_MID   = 512 * MB;
static const size_t S_DW    = 576 * MB;
static const size_t SM      = 640 * MB;
static const size_t W1P  = SM + 0 * MB;
static const size_t W2P  = SM + 4 * MB;
static const size_t W3P  = SM + 5 * MB;
static const size_t F2P  = SM + 9 * MB;
static const size_t F3P  = SM + 10 * MB;
static const size_t STOF = SM + 11 * MB;
static const size_t SEA  = SM + 12 * MB;
static const size_t SEB  = SEA + 1 * MB;
static const size_t SEC  = SEB + 1 * MB;
static const size_t SES1 = SEC + 1 * MB;
static const size_t SES2 = SES1 + 1 * MB;
static const size_t SEG  = SES2 + 1 * MB;

static void gemm(hipStream_t s, const int8_t* A, const int8_t* B, float* C,
                 const float* bias, int M, int N, int Kp)
{
  const int Mt = pad16(M) / 16;
  dim3 grid((unsigned)((Mt + 3) / 4), (unsigned)(pad32i(N) / 32));
  const size_t shbytes = (size_t)32 * Kp;
  k_bingemm<<<grid, 128, shbytes, s>>>(A, B, C, bias, M, N, Kp, Mt);
}

static void bn_act(hipStream_t s, const float* x, const float* st, const float* g,
                   const float* b, float* of, int8_t* os, int M, int C, int Mp,
                   int Kp, int flags)
{
  const size_t tot = (size_t)Mp * Kp;
  k_bn_act<<<(unsigned)((tot + 255) / 256), 256, 0, s>>>(x, st, g, b, of, os,
                                                         M, C, Mp, Kp, flags);
}

struct BIdx {
  int w1, g1, b1, w2, g2, b2, gse1, bse1, fc2w, fc2b, gse2, bse2,
      fc3w, fc3b, gse3, bse3, w3, g3, b3, gres, bres;
};

extern "C" void kernel_launch(void* const* d_in, const int* in_sizes, int n_in,
                              void* d_out, int out_size, void* d_ws, size_t ws_size,
                              hipStream_t stream)
{
  (void)in_sizes; (void)n_in; (void)out_size; (void)ws_size;
  char* wsb = (char*)d_ws;
  auto Fp = [&](size_t off) { return (float*)(wsb + off); };
  auto Sp = [&](size_t off) { return (int8_t*)(wsb + off); };
  auto P  = [&](int i)      { return (const float*)d_in[i]; };

  int pi = 1;
  BIdx B[16];
  for (int i = 0; i < 16; ++i) {
    const Cfg& c = CFGS[i];
    if (c.e != 1) { B[i].w1 = pi++; B[i].g1 = pi++; B[i].b1 = pi++; }
    else          { B[i].w1 = B[i].g1 = B[i].b1 = -1; }
    B[i].w2 = pi++; B[i].g2 = pi++; B[i].b2 = pi++;
    B[i].gse1 = pi++; B[i].bse1 = pi++;
    B[i].fc2w = pi++; B[i].fc2b = pi++;
    B[i].gse2 = pi++; B[i].bse2 = pi++;
    B[i].fc3w = pi++; B[i].fc3b = pi++;
    B[i].gse3 = pi++; B[i].bse3 = pi++;
    B[i].w3 = pi++; B[i].g3 = pi++; B[i].b3 = pi++;
    if (c.s == 1 && c.ci == c.co) { B[i].gres = pi++; B[i].bres = pi++; }
    else                          { B[i].gres = B[i].bres = -1; }
  }
  const int I_conv1 = pi++, I_g1 = pi++, I_b1 = pi++;
  const int I_conv2 = pi++, I_g2 = pi++, I_b2 = pi++;
  const int I_g3 = pi++, I_b3 = pi++;
  const int I_fcw = pi++, I_fcb = pi++;
  const int I_g4 = pi++, I_b4 = pi++;

  const float eps = 1e-3f;

  const int M0 = 32 * 112 * 112;
  {
    const size_t tot = (size_t)M0 * 32;
    k_stem<<<(unsigned)((tot + 255) / 256), 256, 0, stream>>>(
        (const float*)d_in[0], P(I_conv1), Fp(F_GEMM));
    k_bn_stats<<<32, 256, 0, stream>>>(Fp(F_GEMM), Fp(STOF), M0, 32, eps);
    bn_act(stream, Fp(F_GEMM), Fp(STOF), P(I_g1), P(I_b1), Fp(F_X0),
           (int8_t*)nullptr, M0, 32, M0, 32,  1);
  }

  float* cur = Fp(F_X0);
  float* alt = Fp(F_X1);
  int H = 112, W = 112;

  for (int bi = 0; bi < 16; ++bi) {
    const Cfg cg = CFGS[bi];
    const BIdx ix = B[bi];
    const int ci = cg.ci, co = cg.co, kk = cg.k, e = cg.e, st = cg.s;
    const int ce = ci * e, mid = ci / 4;
    const int pdd = (kk - 1) / 2;
    const int Ho = (H + 2 * pdd - kk) / st + 1;
    const int Wo = (W + 2 * pdd - kk) / st + 1;
    const int Min = 32 * H * W, Mout = 32 * Ho * Wo, HWo = Ho * Wo;
    const int KpCi = pad64(ci), KpCe = pad64(ce), KpMid = pad64(mid);
    const int MinP = pad16(Min), MoutP = pad16(Mout);

    {
      const size_t tot = (size_t)MinP * KpCi;
      k_signpack<<<(unsigned)((tot + 255) / 256), 256, 0, stream>>>(
          cur, Sp(S_IN), Min, ci, MinP, KpCi);
    }

    const int8_t* dwin; int dwKp;
    if (e != 1) {
      {
        const size_t tot = (size_t)pad32i(ce) * KpCi;
        k_wpack<<<(unsigned)((tot + 255) / 256), 256, 0, stream>>>(
            P(ix.w1), Sp(W1P), ce, ci, pad32i(ce), KpCi);
      }
      gemm(stream, Sp(S_IN), Sp(W1P), Fp(F_GEMM), nullptr, Min, ce, KpCi);
      k_bn_stats<<<ce, 256, 0, stream>>>(Fp(F_GEMM), Fp(STOF), Min, ce, eps);
      bn_act(stream, Fp(F_GEMM), Fp(STOF), P(ix.g1), P(ix.b1),
             (float*)nullptr, Sp(S_MID), Min, ce, MinP, KpCe, 1);
      dwin = Sp(S_MID); dwKp = KpCe;
    } else {
      dwin = Sp(S_IN); dwKp = KpCi;
    }

    k_sign8<<<(ce * kk * kk + 255) / 256, 256, 0, stream>>>(
        P(ix.w2), Sp(W2P), ce * kk * kk);
    {
      const size_t tot = (size_t)Mout * ce;
      k_dwconv<<<(unsigned)((tot + 255) / 256), 256, 0, stream>>>(
          dwin, Sp(W2P), Fp(F_DWRAW), 32, H, W, ce, dwKp, kk, st, pdd, Ho, Wo);
    }
    k_bn_stats<<<ce, 256, 0, stream>>>(Fp(F_DWRAW), Fp(STOF), Mout, ce, eps);
    bn_act(stream, Fp(F_DWRAW), Fp(STOF), P(ix.g2), P(ix.b2), Fp(F_DWACT),
           (int8_t*)nullptr, Mout, ce, Mout, ce, 1);

    k_avgpool<<<(32 * ce + 255) / 256, 256, 0, stream>>>(
        Fp(F_DWACT), Fp(SEA), 32, HWo, ce);
    k_bn_stats<<<ce, 256, 0, stream>>>(Fp(SEA), Fp(STOF), 32, ce, eps);
    bn_act(stream, Fp(SEA), Fp(STOF), P(ix.gse1), P(ix.bse1),
           (float*)nullptr, Sp(SES1), 32, ce, 32, KpCe, 1);
    {
      const size_t tot = (size_t)pad32i(mid) * KpCe;
      k_wpack<<<(unsigned)((tot + 255) / 256), 256, 0, stream>>>(
          P(ix.fc2w), Sp(F2P), mid, ce, pad32i(mid), KpCe);
    }
    gemm(stream, Sp(SES1), Sp(F2P), Fp(SEB), P(ix.fc2b), 32, mid, KpCe);
    k_bn_stats<<<mid, 256, 0, stream>>>(Fp(SEB), Fp(STOF), 32, mid, eps);
    bn_act(stream, Fp(SEB), Fp(STOF), P(ix.gse2), P(ix.bse2),
           (float*)nullptr, Sp(SES2), 32, mid, 32, KpMid, 1);
    {
      const size_t tot = (size_t)pad32i(ce) * KpMid;
      k_wpack<<<(unsigned)((tot + 255) / 256), 256, 0, stream>>>(
          P(ix.fc3w), Sp(F3P), ce, mid, pad32i(ce), KpMid);
    }
    gemm(stream, Sp(SES2), Sp(F3P), Fp(SEC), P(ix.fc3b), 32, ce, KpMid);
    k_bn_stats<<<ce, 256, 0, stream>>>(Fp(SEC), Fp(STOF), 32, ce, eps);
    bn_act(stream, Fp(SEC), Fp(STOF), P(ix.gse3), P(ix.bse3), Fp(SEG),
           (int8_t*)nullptr, 32, ce, 32, ce,  2);

    {
      const size_t tot = (size_t)MoutP * KpCe;
      k_gate_sign<<<(unsigned)((tot + 255) / 256), 256, 0, stream>>>(
          Fp(F_DWACT), Fp(SEG), Sp(S_DW), Mout, ce, MoutP, KpCe, HWo);
    }

    {
      const size_t tot = (size_t)pad32i(co) * KpCe;
      k_wpack<<<(unsigned)((tot + 255) / 256), 256, 0, stream>>>(
          P(ix.w3), Sp(W3P), co, ce, pad32i(co), KpCe);
    }
    gemm(stream, Sp(S_DW), Sp(W3P), Fp(F_GEMM), nullptr, Mout, co, KpCe);
    k_bn_stats<<<co, 256, 0, stream>>>(Fp(F_GEMM), Fp(STOF), Mout, co, eps);
    if (ix.gres < 0) {
      bn_act(stream, Fp(F_GEMM), Fp(STOF), P(ix.g3), P(ix.b3), alt,
             (int8_t*)nullptr, Mout, co, Mout, co, 1);
    } else {
      bn_act(stream, Fp(F_GEMM), Fp(STOF), P(ix.g3), P(ix.b3), Fp(F_TMP),
             (int8_t*)nullptr, Mout, co, Mout, co, 0);
      const long long n = (long long)Mout * co;
      k_add<<<(unsigned)((n + 255) / 256), 256, 0, stream>>>(
          Fp(F_TMP), cur, Fp(F_TMP), n);
      k_bn_stats<<<co, 256, 0, stream>>>(Fp(F_TMP), Fp(STOF), Mout, co, eps);
      bn_act(stream, Fp(F_TMP), Fp(STOF), P(ix.gres), P(ix.bres), alt,
             (int8_t*)nullptr, Mout, co, Mout, co, 1);
    }
    { float* t = cur; cur = alt; alt = t; }
    H = Ho; W = Wo;
  }

  const int Mh = 32 * H * W;
  const int KpH = pad64(320);
  {
    const size_t tot = (size_t)pad16(Mh) * KpH;
    k_signpack<<<(unsigned)((tot + 255) / 256), 256, 0, stream>>>(
        cur, Sp(S_IN), Mh, 320, pad16(Mh), KpH);
  }
  {
    const size_t tot = (size_t)1280 * KpH;
    k_wpack<<<(unsigned)((tot + 255) / 256), 256, 0, stream>>>(
        P(I_conv2), Sp(W1P), 1280, 320, 1280, KpH);
  }
  gemm(stream, Sp(S_IN), Sp(W1P), Fp(F_GEMM), nullptr, Mh, 1280, KpH);
  k_bn_stats<<<1280, 256, 0, stream>>>(Fp(F_GEMM), Fp(STOF), Mh, 1280, eps);
  bn_act(stream, Fp(F_GEMM), Fp(STOF), P(I_g2), P(I_b2), Fp(F_DWACT),
         (int8_t*)nullptr, Mh, 1280, Mh, 1280, 1);
  k_avgpool<<<(32 * 1280 + 255) / 256, 256, 0, stream>>>(
      Fp(F_DWACT), Fp(SEA), 32, H * W, 1280);
  k_bn_stats<<<1280, 256, 0, stream>>>(Fp(SEA), Fp(STOF), 32, 1280, eps);
  bn_act(stream, Fp(SEA), Fp(STOF), P(I_g3), P(I_b3), (float*)nullptr,
         Sp(SES1), 32, 1280, 32, 1280, 1);
  {
    const size_t tot = (size_t)pad32i(1000) * 1280;
    k_wpack<<<(unsigned)((tot + 255) / 256), 256, 0, stream>>>(
        P(I_fcw), Sp(W3P), 1000, 1280, pad32i(1000), 1280);
  }
  gemm(stream, Sp(SES1), Sp(W3P), Fp(SEB), P(I_fcb), 32, 1000, 1280);
  k_bn_stats<<<1000, 256, 0, stream>>>(Fp(SEB), Fp(STOF), 32, 1000, eps);
  bn_act(stream, Fp(SEB), Fp(STOF), P(I_g4), P(I_b4), Fp(SEC),
         (int8_t*)nullptr, 32, 1000, 32, 1000, 0);
  k_logsoftmax<<<32, 256, 0, stream>>>(Fp(SEC), (float*)d_out, 1000);
}
